// NVPModel_10496900072142
// MI455X (gfx1250) — matrix operand order repaired and checked by reading, not compiled, not run on hardware
//
#include <hip/hip_runtime.h>

typedef __attribute__((ext_vector_type(16))) _Float16 v16h;
typedef __attribute__((ext_vector_type(8)))  float    v8f;

#define NV_SLOTS 40                       // 8 nets x 5 layers
#define WPACK_HALFS (NV_SLOTS * 32 * 16)  // 20480 f16 = 40960 B

struct NVPParams {
  const float* W[NV_SLOTS];
  const float* b[NV_SLOTS];
};

__device__ __forceinline__ void ldsfence() {
  // Same-wave DS ops are processed in-order by the LDS unit (CDNA5 ISA);
  // we only need to stop compile-time reordering of LDS traffic.
  __builtin_amdgcn_wave_barrier();
  asm volatile("" ::: "memory");
}

__device__ __forceinline__ float fast_tanh(float x) {
#if __has_builtin(__builtin_amdgcn_tanhf)
  return __builtin_amdgcn_tanhf(x);   // V_TANH_F32 (CDNA5 TRANS op)
#else
  float cx = fminf(fmaxf(x, -12.f), 12.f);
  float e = __expf(2.f * cx);
  return (e - 1.f) / (e + 1.f);
#endif
}

// ReLU as a single v_max_i32: positive floats compare as positive ints,
// all negative floats (incl -0) compare as negative ints.
__device__ __forceinline__ float relu_f(float v) {
  int b = __float_as_int(v);
  b = (b > 0) ? b : 0;
  return __int_as_float(b);
}

// ---------------- weight pre-pack ------------------------------------------
// B-matrix (32x16, f16) fragment per wave32 lane l, half j (0..15):
//   element B[k][n], n = l&15, k = (l>>4)*8 + j for j<8 (j>=8: K=16..31, zero).
// Bias is folded into weight row k=15 (homogeneous trick): the A fragment
// carries a constant 1.0 at k=15, so D = W*h + b with an inline-0 C operand.
__global__ void nvp_prep(NVPParams P, _Float16* wsW) {
  int tid = blockIdx.x * blockDim.x + threadIdx.x;
  if (tid >= NV_SLOTS * 32) return;
  int slot = tid >> 5, lane = tid & 31;
  int layer = slot % 5;
  int Kr = (layer == 0) ? 8 : 10;
  int Nr = (layer == 4) ? 8 : 10;
  const float* W = P.W[slot];  // (fan_in, fan_out) row-major: W[k*Nr + n]
  const float* b = P.b[slot];
  int n = lane & 15;
  _Float16* dst = wsW + (size_t)(slot * 32 + lane) * 16;
  for (int j = 0; j < 16; ++j) {
    float v = 0.f;
    int k = (lane >> 4) * 8 + j;
    if (j < 8 && n < Nr) {
      if (k < Kr) v = W[k * Nr + n];
      else if (k == 15) v = b[n];  // bias row
    }
    dst[j] = (_Float16)v;
  }
}

// ---------------- A-fragment builder ---------------------------------------
// A (16x32 f16): lane l holds row m=l&15; halves j<8 carry K=(l>>4)*8+j.
// Halves 8..15 (K=16..31) are zeroed ONCE at init and never touched again.
// Scratch-tile column 15 permanently holds 1.0 (homogeneous bias input);
// columns N_real..14 are always zero, so one builder serves every layer.
__device__ __forceinline__ void build_a_low(v16h& a, const float* t, int lane) {
  const int m = lane & 15, k0 = (lane >> 4) * 8;
#pragma unroll
  for (int j = 0; j < 8; ++j) a[j] = (_Float16)t[m * 16 + k0 + j];
}

// ---------------- main kernel ----------------------------------------------
// 128 threads = 4 waves; each wave owns TWO 16-sample tiles so that every
// B-fragment load feeds two WMMAs and four independent WMMA chains
// (s/t x tile0/tile1) fill the WMMA->VALU hazard slots.
__global__ void __launch_bounds__(128)
nvp_main(const float* __restrict__ x, const _Float16* __restrict__ gW,
         float* __restrict__ outH, float* __restrict__ outLP, int nTiles) {
  __shared__ __align__(32) _Float16 sW[WPACK_HALFS];   // 40 KB
  __shared__ __align__(16) float sT[4][4][16 * 16];    // 16 KB: 4 tiles/wave

  const int tid = threadIdx.x;
  const int w = tid >> 5, lane = tid & 31;

  // stage packed weights into LDS (shared by all 4 waves): 2560 uint4
  {
    const uint4* src = (const uint4*)gW;
    uint4* dst = (uint4*)sW;
#pragma unroll
    for (int i = 0; i < 20; ++i) dst[tid + i * 128] = src[tid + i * 128];
  }
  __syncthreads();

  const int pair = blockIdx.x * 4 + w;  // this wave's 2-tile group
  if (pair * 2 >= nTiles) return;
  const long long s0 = (long long)pair * 32;  // 32 consecutive samples

  // scratch: [0]=s-chain tile0, [1]=t-chain tile0, [2]=s tile1, [3]=t tile1
  float* tb0 = &sT[w][0][0];
  float* tb1 = &sT[w][1][0];
  float* tb2 = &sT[w][2][0];
  float* tb3 = &sT[w][3][0];
  const int n = lane & 15, mo = (lane >> 4) * 8;

  // constant homogeneous column: tb[:,15] = 1.0 for the whole kernel
  if (lane < 16) {
    tb0[lane * 16 + 15] = 1.0f;
    tb1[lane * 16 + 15] = 1.0f;
    tb2[lane * 16 + 15] = 1.0f;
    tb3[lane * 16 + 15] = 1.0f;
  }
  ldsfence();

  // h1/h2 in registers, WMMA C/D layout; [u] = tile index
  float H1[2][8], H2[2][8], acc[2][8];
#pragma unroll
  for (int u = 0; u < 2; ++u) {
    if (n < 8) {
#pragma unroll
      for (int r = 0; r < 8; ++r) {
        const float2 xv =
            *(const float2*)(x + (s0 + u * 16 + mo + r) * 16 + 2 * n);
        H1[u][r] = xv.x;
        H2[u][r] = xv.y;
      }
    } else {
#pragma unroll
      for (int r = 0; r < 8; ++r) { H1[u][r] = 0.f; H2[u][r] = 0.f; }
    }
#pragma unroll
    for (int r = 0; r < 8; ++r) acc[u][r] = 0.f;
  }

  // persistent A fragments: high halves (K=16..31) zeroed once
  v16h A0, A1, A2, A3;
#pragma unroll
  for (int j = 0; j < 16; ++j) {
    A0[j] = (_Float16)0.f; A1[j] = (_Float16)0.f;
    A2[j] = (_Float16)0.f; A3[j] = (_Float16)0.f;
  }

#pragma unroll
  for (int c = 0; c < 4; ++c) {
    ldsfence();
    // stage the source half of each tile (col 15 keeps its constant 1.0)
    if (n != 15) {
#pragma unroll
      for (int r = 0; r < 8; ++r) {
        tb0[(mo + r) * 16 + n] = (c & 1) ? H2[0][r] : H1[0][r];
        tb2[(mo + r) * 16 + n] = (c & 1) ? H2[1][r] : H1[1][r];
      }
    }
    ldsfence();
    build_a_low(A0, tb0, lane);
    build_a_low(A2, tb2, lane);
#pragma unroll
    for (int j = 0; j < 8; ++j) { A1[j] = A0[j]; A3[j] = A2[j]; }
    ldsfence();

    v8f D0, D1, D2, D3;
#pragma unroll
    for (int layer = 0; layer < 5; ++layer) {
      const int ss = c * 5 + layer;        // s-net: nets 0..3
      const int st = (4 + c) * 5 + layer;  // t-net: nets 4..7
      const v16h Bs = *(const v16h*)&sW[(size_t)(ss * 32 + lane) * 16];
      const v16h Bt = *(const v16h*)&sW[(size_t)(st * 32 + lane) * 16];
      v8f z = {};  // inline-0 C operand (bias folded into W row 15)
      D0 = __builtin_amdgcn_wmma_f32_16x16x32_f16(false, A0, false, Bs,
                                                  (short)0, z, false, false);
      D1 = __builtin_amdgcn_wmma_f32_16x16x32_f16(false, A1, false, Bt,
                                                  (short)0, z, false, false);
      D2 = __builtin_amdgcn_wmma_f32_16x16x32_f16(false, A2, false, Bs,
                                                  (short)0, z, false, false);
      D3 = __builtin_amdgcn_wmma_f32_16x16x32_f16(false, A3, false, Bt,
                                                  (short)0, z, false, false);
      if (layer < 4) {
        if (n != 15) {
#pragma unroll
          for (int r = 0; r < 8; ++r) {
            tb0[(mo + r) * 16 + n] = relu_f(D0[r]);
            tb1[(mo + r) * 16 + n] = relu_f(D1[r]);
            tb2[(mo + r) * 16 + n] = relu_f(D2[r]);
            tb3[(mo + r) * 16 + n] = relu_f(D3[r]);
          }
        }
        ldsfence();
        build_a_low(A0, tb0, lane);
        build_a_low(A1, tb1, lane);
        build_a_low(A2, tb2, lane);
        build_a_low(A3, tb3, lane);
        ldsfence();
      }
    }

    // coupling update, all in registers (padded columns give tanh(0)=0)
#pragma unroll
    for (int u = 0; u < 2; ++u) {
      const v8f& Dsu = u ? D2 : D0;
      const v8f& Dtu = u ? D3 : D1;
      float stv[8];
#pragma unroll
      for (int r = 0; r < 8; ++r) {
        stv[r] = 3.f * fast_tanh(Dsu[r]);  // == log(scale)
        acc[u][r] += stv[r];
      }
      if (n < 8) {
#pragma unroll
        for (int r = 0; r < 8; ++r) {
          const float sc = __expf(stv[r]);
          const float tt = 3.f * Dtu[r];
          if (c & 1) H1[u][r] = H1[u][r] * sc + tt;
          else       H2[u][r] = H2[u][r] * sc + tt;
        }
      }
    }
  }

  // write interleaved h straight from registers
#pragma unroll
  for (int u = 0; u < 2; ++u) {
    if (n < 8) {
#pragma unroll
      for (int r = 0; r < 8; ++r) {
        float2 o;
        o.x = H1[u][r];
        o.y = H2[u][r];
        *(float2*)(outH + (s0 + u * 16 + mo + r) * 16 + 2 * n) = o;
      }
    }
    // logph: reduce over feature columns within each 16-lane half
    float red[8];
#pragma unroll
    for (int r = 0; r < 8; ++r) {
      float v = acc[u][r];
      v += __shfl_xor(v, 1, 16);
      v += __shfl_xor(v, 2, 16);
      v += __shfl_xor(v, 4, 16);
      v += __shfl_xor(v, 8, 16);
      red[r] = v;
    }
    if ((lane & 15) == 0) {
      float4* lp = (float4*)(outLP + s0 + u * 16 + mo);
      float4 l0, l1;
      l0.x = red[0]; l0.y = red[1]; l0.z = red[2]; l0.w = red[3];
      l1.x = red[4]; l1.y = red[5]; l1.z = red[6]; l1.w = red[7];
      lp[0] = l0;
      lp[1] = l1;
    }
  }
}

// ---------------- host launch ----------------------------------------------
extern "C" void kernel_launch(void* const* d_in, const int* in_sizes, int n_in,
                              void* d_out, int out_size, void* d_ws,
                              size_t ws_size, hipStream_t stream) {
  // locate x = the big input; remaining 80 leaves are param arrays in order
  int xi = 0;
  for (int i = 1; i < n_in; ++i)
    if (in_sizes[i] > in_sizes[xi]) xi = i;
  const float* x = (const float*)d_in[xi];
  long long B = (long long)in_sizes[xi] / 16;

  int pidx[80];
  int c = 0;
  for (int i = 0; i < n_in && c < 80; ++i)
    if (i != xi) pidx[c++] = i;

  // interleaved (W1,b1,W2,b2,...) => 2nd leaf of a net has size 10;
  // sorted (W1..W5,b1..b5)        => 2nd leaf has size 100.
  bool interleaved = (in_sizes[pidx[1]] <= 16);

  NVPParams P;
  for (int net = 0; net < 8; ++net) {
    for (int l = 0; l < 5; ++l) {
      int wi, bi;
      if (interleaved) {
        wi = pidx[net * 10 + 2 * l];
        bi = pidx[net * 10 + 2 * l + 1];
      } else {
        wi = pidx[net * 10 + l];
        bi = pidx[net * 10 + 5 + l];
      }
      P.W[net * 5 + l] = (const float*)d_in[wi];
      P.b[net * 5 + l] = (const float*)d_in[bi];
    }
  }

  _Float16* wsW = (_Float16*)d_ws;
  nvp_prep<<<5, 256, 0, stream>>>(P, wsW);

  int nTiles = (int)(B / 16);
  // each block: 4 waves x 2 tiles = 8 tiles
  int blocks = (nTiles + 7) / 8;
  float* outH = (float*)d_out;
  float* outLP = outH + (size_t)B * 16;
  nvp_main<<<blocks, 128, 0, stream>>>(x, wsW, outH, outLP, nTiles);
}
